// SVD_58317065945587
// MI455X (gfx1250) — matrix operand order repaired and checked by reading, not compiled, not run on hardware
//
#include <hip/hip_runtime.h>
#include <math.h>
#include <stdint.h>

#define BATCH 16
#define NPTS  2048
#define DIM   512
#define BQ    128   // queries per block (8 waves * 16)
#define KT    32    // keys per tile (two 16-key WMMA sub-tiles)
#define NTILES (NPTS / KT)          // 64
#define TILE_ELEMS (KT * DIM)       // 16384 bf16 = 32 KB

typedef __attribute__((ext_vector_type(16))) __bf16 bf16x16;
typedef __attribute__((ext_vector_type(8)))  __bf16 bf16x8;
typedef __attribute__((ext_vector_type(4)))  __bf16 bf16x4;
typedef __attribute__((ext_vector_type(8)))  float  f32x8;

// ---------------------------------------------------------------------------
// Kernel 1: f32 -> bf16 conversion of both embeddings.
// src_embed (queries) gets the 1/d_k = 1/3 scale folded in.
// ---------------------------------------------------------------------------
__global__ __launch_bounds__(256)
void cvt_kernel(const float4* __restrict__ sE, const float4* __restrict__ tE,
                bf16x4* __restrict__ qb, bf16x4* __restrict__ kb, int total4) {
    const float inv3 = 1.0f / 3.0f;
    for (int i = blockIdx.x * blockDim.x + threadIdx.x; i < total4;
         i += gridDim.x * blockDim.x) {
        float4 a = sE[i];
        float4 t = tE[i];
        bf16x4 qa, ka;
        qa[0] = (__bf16)(a.x * inv3); qa[1] = (__bf16)(a.y * inv3);
        qa[2] = (__bf16)(a.z * inv3); qa[3] = (__bf16)(a.w * inv3);
        ka[0] = (__bf16)t.x; ka[1] = (__bf16)t.y;
        ka[2] = (__bf16)t.z; ka[3] = (__bf16)t.w;
        qb[i] = qa;
        kb[i] = ka;
    }
}

// ---------------------------------------------------------------------------
// Kernel 2: flash attention with v_wmma_f32_16x16x32_bf16.
// Key tiles are DMA'd into double-buffered LDS with
// global_load_async_to_lds_b128 (ASYNCcnt), one barrier per tile, copy
// overlapped with compute. Q operands persistent in VGPRs.
// ---------------------------------------------------------------------------
__global__ __launch_bounds__(256)
void attn_kernel(const __bf16* __restrict__ qbf, const __bf16* __restrict__ kbf,
                 const float* __restrict__ tgt, float* __restrict__ matched) {
    const int b    = blockIdx.y;
    const int lane = threadIdx.x & 31;
    const int wv   = threadIdx.x >> 5;
    const int half = lane >> 4;     // which 16-lane half of the wave
    const int l15  = lane & 15;
    const int qr   = blockIdx.x * BQ + wv * 16 + l15;  // this lane's query

    __shared__ __bf16 kt_lds[2][TILE_ELEMS];

    // Preload Q as WMMA B-operands: lane holds column q = qr,
    // halves 0..15 = K (c*32 + half*16 + 0..15)  (contiguous 32B per chunk).
    bf16x16 bq[16];
    {
        const __bf16* qrow = qbf + ((size_t)b * NPTS + qr) * DIM + half * 16;
#pragma unroll
        for (int c = 0; c < 16; ++c)
            bq[c] = *(const bf16x16*)(qrow + c * 32);
    }

    // Flash-softmax running state (per query column; each half accumulates
    // its own 8-of-16 keys per sub-tile, merged at the end).
    float run_max = -INFINITY, run_sum = 0.0f;
    float o0 = 0.0f, o1 = 0.0f, o2 = 0.0f;

    // Async staging: each thread copies 128 contiguous bytes of the
    // (32 x 512 bf16) tile: 8 x b128, one shared offset immediate per chunk
    // (the inst offset adds to BOTH the global and LDS address).
    const int t = threadIdx.x;
    const __bf16* kstage = kbf + ((size_t)b * NPTS) * DIM + (size_t)t * 64;
    const uint32_t loff0 = (uint32_t)(uintptr_t)(&kt_lds[0][0] + t * 64);
    const uint32_t loff1 = loff0 + TILE_ELEMS * 2;  // bytes

    // prologue: kick off tile 0 into buffer 0
    {
        const __bf16* g = kstage;
#pragma unroll
        for (int i = 0; i < 8; ++i)
            asm volatile("global_load_async_to_lds_b128 %0, %1, off offset:%2"
                         :: "v"(loff0), "v"(g), "i"(i * 16) : "memory");
    }

    const float* vbase = tgt + (size_t)b * NPTS * 3;

    for (int mt = 0; mt < NTILES; ++mt) {
        // own wave's DMA for tile mt complete:
        asm volatile("s_wait_asynccnt 0x0" ::: "memory");
        // all waves' DMA visible + all waves done reading buf[(mt+1)&1]:
        __syncthreads();

        if (mt + 1 < NTILES) {  // uniform branch
            const uint32_t loff = ((mt + 1) & 1) ? loff1 : loff0;
            const __bf16* g = kstage + (size_t)(mt + 1) * TILE_ELEMS;
#pragma unroll
            for (int i = 0; i < 8; ++i)
                asm volatile("global_load_async_to_lds_b128 %0, %1, off offset:%2"
                             :: "v"(loff), "v"(g), "i"(i * 16) : "memory");
        }

        const __bf16* tile = kt_lds[mt & 1];
        const __bf16* arow0 = tile + l15 * DIM + half * 16;       // keys +0..15
        const __bf16* arow1 = arow0 + 16 * DIM;                   // keys +16..31

        // S sub-tile 0: acc0[r] = S[key = mt*32 + half*8 + r][query qr]
        f32x8 acc0 = {0.f, 0.f, 0.f, 0.f, 0.f, 0.f, 0.f, 0.f};
#pragma unroll
        for (int c = 0; c < 16; ++c) {
            // A operand: halves 0..15 = K kb+half*16+0..15 (contiguous, matches bq)
            bf16x8 lo = *(const bf16x8*)(arow0 + c * 32);
            bf16x8 hi = *(const bf16x8*)(arow0 + c * 32 + 8);
            bf16x16 a = __builtin_shufflevector(lo, hi, 0, 1, 2, 3, 4, 5, 6, 7,
                                                8, 9, 10, 11, 12, 13, 14, 15);
            acc0 = __builtin_amdgcn_wmma_f32_16x16x32_bf16(
                false, a, false, bq[c], (short)0, acc0, false, false);
        }
        // S sub-tile 1: acc1[r] = S[key = mt*32 + 16 + half*8 + r][query qr]
        f32x8 acc1 = {0.f, 0.f, 0.f, 0.f, 0.f, 0.f, 0.f, 0.f};
#pragma unroll
        for (int c = 0; c < 16; ++c) {
            bf16x8 lo = *(const bf16x8*)(arow1 + c * 32);
            bf16x8 hi = *(const bf16x8*)(arow1 + c * 32 + 8);
            bf16x16 a = __builtin_shufflevector(lo, hi, 0, 1, 2, 3, 4, 5, 6, 7,
                                                8, 9, 10, 11, 12, 13, 14, 15);
            acc1 = __builtin_amdgcn_wmma_f32_16x16x32_bf16(
                false, a, false, bq[c], (short)0, acc1, false, false);
        }

        // online softmax over the 32 keys of this tile
        float tmax = fmaxf(acc0[0], acc1[0]);
#pragma unroll
        for (int r = 1; r < 8; ++r)
            tmax = fmaxf(tmax, fmaxf(acc0[r], acc1[r]));
        tmax = fmaxf(tmax, __shfl_xor(tmax, 16, 32));  // both halves agree
        float nmax = fmaxf(run_max, tmax);
        float corr = __expf(run_max - nmax);
        run_max = nmax;
        run_sum *= corr; o0 *= corr; o1 *= corr; o2 *= corr;

        const float* vrow0 = vbase + (size_t)(mt * KT + half * 8) * 3;
        const float* vrow1 = vrow0 + 16 * 3;
#pragma unroll
        for (int r = 0; r < 8; ++r) {
            float p = __expf(acc0[r] - nmax);
            run_sum += p;
            o0 = fmaf(p, vrow0[r * 3 + 0], o0);
            o1 = fmaf(p, vrow0[r * 3 + 1], o1);
            o2 = fmaf(p, vrow0[r * 3 + 2], o2);
            float q = __expf(acc1[r] - nmax);
            run_sum += q;
            o0 = fmaf(q, vrow1[r * 3 + 0], o0);
            o1 = fmaf(q, vrow1[r * 3 + 1], o1);
            o2 = fmaf(q, vrow1[r * 3 + 2], o2);
        }
    }

    // merge the two lane halves (each covered half of the keys)
    float tot = run_sum + __shfl_xor(run_sum, 16, 32);
    float f0  = o0 + __shfl_xor(o0, 16, 32);
    float f1  = o1 + __shfl_xor(o1, 16, 32);
    float f2  = o2 + __shfl_xor(o2, 16, 32);
    float inv = 1.0f / tot;
    if (half == 0) {
        float* mout = matched + ((size_t)b * NPTS + qr) * 3;
        mout[0] = f0 * inv;
        mout[1] = f1 * inv;
        mout[2] = f2 * inv;
    }
}

// ---------------------------------------------------------------------------
// Kernel 3: per-batch reduction + 3x3 SVD (via Jacobi on H^T H) -> R, t.
// Deterministic tree reduction (no float atomics).
// ---------------------------------------------------------------------------
__global__ __launch_bounds__(256)
void finish_kernel(const float* __restrict__ src, const float* __restrict__ matched,
                   float* __restrict__ out) {
    const int b = blockIdx.x;
    const int t = threadIdx.x;
    float v[15];
#pragma unroll
    for (int k = 0; k < 15; ++k) v[k] = 0.0f;

    for (int n = t; n < NPTS; n += 256) {
        const float* sp = src + ((size_t)b * NPTS + n) * 3;
        const float* mp = matched + ((size_t)b * NPTS + n) * 3;
        float s0 = sp[0], s1 = sp[1], s2 = sp[2];
        float m0 = mp[0], m1 = mp[1], m2 = mp[2];
        v[0] += s0; v[1] += s1; v[2] += s2;
        v[3] += m0; v[4] += m1; v[5] += m2;
        v[6]  += s0 * m0; v[7]  += s0 * m1; v[8]  += s0 * m2;
        v[9]  += s1 * m0; v[10] += s1 * m1; v[11] += s1 * m2;
        v[12] += s2 * m0; v[13] += s2 * m1; v[14] += s2 * m2;
    }

    __shared__ float red[8][15];
    const int lane = t & 31, wv = t >> 5;
#pragma unroll
    for (int k = 0; k < 15; ++k) {
        float x = v[k];
        for (int off = 16; off > 0; off >>= 1) x += __shfl_down(x, off, 32);
        if (lane == 0) red[wv][k] = x;
    }
    __syncthreads();

    if (t == 0) {
        float tot[15];
#pragma unroll
        for (int k = 0; k < 15; ++k) {
            float a = 0.0f;
            for (int w = 0; w < 8; ++w) a += red[w][k];
            tot[k] = a;
        }
        const float invN = 1.0f / (float)NPTS;
        float sb[3] = {tot[0] * invN, tot[1] * invN, tot[2] * invN};
        float mb[3] = {tot[3] * invN, tot[4] * invN, tot[5] * invN};
        float H[3][3];
        for (int i = 0; i < 3; ++i)
            for (int j = 0; j < 3; ++j)
                H[i][j] = tot[6 + i * 3 + j] - (float)NPTS * sb[i] * mb[j];

        // A = H^T H (symmetric), Jacobi eigendecomposition A = V L V^T
        float A[3][3], V[3][3] = {{1, 0, 0}, {0, 1, 0}, {0, 0, 1}};
        for (int i = 0; i < 3; ++i)
            for (int j = 0; j < 3; ++j) {
                float a = 0.0f;
                for (int k = 0; k < 3; ++k) a += H[k][i] * H[k][j];
                A[i][j] = a;
            }
        const int PP[3] = {0, 0, 1}, QQ[3] = {1, 2, 2};
        for (int sweep = 0; sweep < 12; ++sweep) {
            for (int r = 0; r < 3; ++r) {
                int p = PP[r], q = QQ[r];
                float apq = A[p][q];
                if (fabsf(apq) > 1e-20f) {
                    float tau = (A[q][q] - A[p][p]) / (2.0f * apq);
                    float tt = (tau >= 0.0f ? 1.0f : -1.0f) /
                               (fabsf(tau) + sqrtf(1.0f + tau * tau));
                    float c = 1.0f / sqrtf(1.0f + tt * tt);
                    float s = tt * c;
                    for (int k = 0; k < 3; ++k) {  // A <- A*J
                        float akp = A[k][p], akq = A[k][q];
                        A[k][p] = c * akp - s * akq;
                        A[k][q] = s * akp + c * akq;
                    }
                    for (int k = 0; k < 3; ++k) {  // A <- J^T*A
                        float apk = A[p][k], aqk = A[q][k];
                        A[p][k] = c * apk - s * aqk;
                        A[q][k] = s * apk + c * aqk;
                    }
                    for (int k = 0; k < 3; ++k) {  // V <- V*J
                        float vkp = V[k][p], vkq = V[k][q];
                        V[k][p] = c * vkp - s * vkq;
                        V[k][q] = s * vkp + c * vkq;
                    }
                }
            }
        }
        // W = V diag(1/sqrt(l)) V^T = (H^T H)^(-1/2)
        float is[3];
        for (int i = 0; i < 3; ++i) is[i] = rsqrtf(fmaxf(A[i][i], 1e-12f));
        float W[3][3];
        for (int i = 0; i < 3; ++i)
            for (int j = 0; j < 3; ++j) {
                float a = 0.0f;
                for (int k = 0; k < 3; ++k) a += V[i][k] * is[k] * V[j][k];
                W[i][j] = a;
            }
        // R = (H * W)^T  (= V U^T of the reference)
        float R[3][3];
        for (int i = 0; i < 3; ++i)
            for (int j = 0; j < 3; ++j) {
                float a = 0.0f;
                for (int k = 0; k < 3; ++k) a += H[j][k] * W[k][i];
                R[i][j] = a;
            }
        float det = R[0][0] * (R[1][1] * R[2][2] - R[1][2] * R[2][1]) -
                    R[0][1] * (R[1][0] * R[2][2] - R[1][2] * R[2][0]) +
                    R[0][2] * (R[1][0] * R[2][1] - R[1][1] * R[2][0]);
        if (det < 0.0f) {
            R[2][0] = -R[2][0]; R[2][1] = -R[2][1]; R[2][2] = -R[2][2];
        }
        for (int i = 0; i < 3; ++i)
            for (int j = 0; j < 3; ++j) out[b * 9 + i * 3 + j] = R[i][j];
        for (int j = 0; j < 3; ++j)
            out[BATCH * 9 + b * 3 + j] =
                mb[j] - (sb[0] * R[j][0] + sb[1] * R[j][1] + sb[2] * R[j][2]);
    }
}

// ---------------------------------------------------------------------------
extern "C" void kernel_launch(void* const* d_in, const int* in_sizes, int n_in,
                              void* d_out, int out_size, void* d_ws, size_t ws_size,
                              hipStream_t stream) {
    const float* src  = (const float*)d_in[0];  // [B,N,3]
    const float* tgt  = (const float*)d_in[1];  // [B,N,3]
    const float* srcE = (const float*)d_in[2];  // [B,N,D]  (queries)
    const float* tgtE = (const float*)d_in[3];  // [B,N,D]  (keys)
    float* out = (float*)d_out;                 // R[16,3,3] ++ t[16,1,3]

    const size_t nElems  = (size_t)BATCH * NPTS * DIM;   // 16.78M
    const size_t qkBytes = nElems * sizeof(__bf16);      // 33.5 MB each
    const size_t need    = 2 * qkBytes + (size_t)BATCH * NPTS * 3 * sizeof(float);
    if (ws_size < need) return;  // scratch too small; nothing safe to do

    char* ws = (char*)d_ws;
    __bf16* qbf    = (__bf16*)ws;
    __bf16* kbf    = (__bf16*)(ws + qkBytes);
    float* matched = (float*)(ws + 2 * qkBytes);

    const int total4 = (int)(nElems / 4);
    cvt_kernel<<<4096, 256, 0, stream>>>((const float4*)srcE, (const float4*)tgtE,
                                         (bf16x4*)qbf, (bf16x4*)kbf, total4);

    dim3 grid(NPTS / BQ, BATCH);  // 16 x 16 = 256 blocks, 8 waves each
    attn_kernel<<<grid, 256, 0, stream>>>(qbf, kbf, tgt, matched);

    finish_kernel<<<BATCH, 256, 0, stream>>>(src, matched, out);
}
